// langModel_88751204205228
// MI455X (gfx1250) — matrix operand order repaired and checked by reading, not compiled, not run on hardware
//
#include <hip/hip_runtime.h>
#include <hip/hip_bf16.h>
#include <math.h>

typedef __attribute__((ext_vector_type(16))) __bf16 v16bf;
typedef __attribute__((ext_vector_type(8)))  float  v8f;

// Problem constants
constexpr int Bc = 64;
constexpr int Tc = 1024;
constexpr int Hc = 512;
constexpr int Lc = 2;
constexpr int NG = 4 * Hc;                            // 2048 gate columns
constexpr size_t CELL_W_HALFS = (size_t)NG * 2 * Hc;  // 2,097,152 bf16 per cell
constexpr size_t APACK_HALFS  = (size_t)Bc * Hc;      // 32768 halfs = 64 KiB per act

// ---------------------------------------------------------------------------
// Pack W_ih|W_hh (fp32, row-major (4H,H)) into bf16 WMMA-B tiles.
// Layout: Wp[cell][tile(=n/16, 0..127)][kc(0..31)][lane(0..31)][j(0..15)]
// element = Wcat[k = kc*32 + (lane>=16?8:0)+(j<8?j:j+8), n = tile*16 + (lane&15)],
// Wcat[k,n] = (k<512) ? W_ih[n,k] : W_hh[n,k-512].
// Each (tile,kc) chunk is 1 KiB; a lane's 16 halfs are 32 contiguous bytes.
// ---------------------------------------------------------------------------
__global__ __launch_bounds__(256) void pack_weights(
    const float* __restrict__ Wih, const float* __restrict__ Whh,
    __bf16* __restrict__ Wp)
{
    size_t idx = (size_t)blockIdx.x * 256 + threadIdx.x;  // 8,388,608 total
    int j    = (int)(idx & 15);
    int lane = (int)((idx >> 4) & 31);
    int kc   = (int)((idx >> 9) & 31);
    int tile = (int)((idx >> 14) & 127);
    int cell = (int)(idx >> 21);
    if (cell >= 4) return;
    int k = kc * 32 + ((lane >> 4) << 3) + (j < 8 ? j : j + 8);
    int n = tile * 16 + (lane & 15);
    size_t wbase = (size_t)cell * NG * Hc;   // cell = l*2+d, matches (L,2,4H,H)
    float v = (k < Hc) ? Wih[wbase + (size_t)n * Hc + k]
                       : Whh[wbase + (size_t)n * Hc + (k - Hc)];
    size_t off = ((((size_t)cell * 128 + tile) * 32 + kc) * 32 + lane) * 16 + j;
    Wp[off] = (__bf16)v;
}

__global__ __launch_bounds__(256) void pack_bias(
    const float* __restrict__ bih, const float* __restrict__ bhh,
    float* __restrict__ bout)
{
    int i = blockIdx.x * 256 + threadIdx.x;
    if (i < 4 * NG) bout[i] = bih[i] + bhh[i];
}

__global__ __launch_bounds__(256) void zero_f32(float* __restrict__ p, int n)
{
    int i = blockIdx.x * 256 + threadIdx.x;
    if (i < n) p[i] = 0.f;
}

// ---------------------------------------------------------------------------
// Fused LSTM cell: g = x@Wih^T + h@Whh^T + b ; gates ; state update.
// Grid (32,4): blockIdx.x = n-tile (h columns), blockIdx.y = m-tile (batch).
// Block 256 = 8 waves: wave w -> gate = w&3, khalf = w>>2 (0: x K-half,
// 1: h K-half). Each wave runs a 16-deep v_wmma_f32_16x16x32_bf16 chain,
// halves reduced through LDS, then 256 threads do the elementwise update.
//
// A operands come pre-packed in bf16 WMMA-A register layout
// (pack[((mt*16+kc)*32+lane)*16+j], k = hi*8 + (j<8?j:j+8) + kc*32,
//  m = mt*16 + (lane&15)): two b128 loads per k-step, no cvt. A null pack
// pointer (raw fp32 X input) selects a separate fp32+cvt loop. The selector
// is wave-uniform, so it is forced into an SGPR via readfirstlane and the
// branch is a scalar s_cbranch OUTSIDE the k-loop (EXEC stays all-ones for
// WMMA; no per-iteration exec juggling).
// c == nullptr means c_prev = 0.
// The elementwise stage also emits the bf16 A-pack of oh for the next cell.
// ---------------------------------------------------------------------------
__global__ __launch_bounds__(256) void lstm_cell(
    const __bf16* __restrict__ xp, const float* __restrict__ xf, long xfs,
    const __bf16* __restrict__ hp, const float* __restrict__ hf, long hfs,
    const float* __restrict__ c,  long cs,
    float* __restrict__ oh, long ohs,
    float* __restrict__ oc, long ocs,
    __bf16* __restrict__ ohp,
    const __bf16* __restrict__ Wp,
    const float* __restrict__ bias)
{
    __shared__ float lds[2][4][16][16];   // [khalf][gate][m][n], 8 KiB

    const int tid   = threadIdx.x;
    const int lane  = tid & 31;
    const int w     = tid >> 5;           // 0..7
    const int gate  = w & 3;
    const int khalf = w >> 2;             // 0 -> x, 1 -> h
    const int nt    = blockIdx.x;         // 0..31
    const int mt    = blockIdx.y;         // 0..3
    const int hi    = lane >> 4;          // lane group (ISA A/B layouts)

    const __bf16* apack = khalf ? hp  : xp;
    const float*  afp   = khalf ? hf  : xf;
    const long    afs   = khalf ? hfs : xfs;

    const int tile = gate * 32 + nt;      // 0..127 global n-tile in g
    const char* wtile = (const char*)Wp
        + ((size_t)tile * 32 + (size_t)khalf * 16) * 1024; // 1 KiB per kc chunk

    // Wave-uniform path selector -> SGPR -> scalar branch
    const int usepk = __builtin_amdgcn_readfirstlane(apack != nullptr ? 1 : 0);

    v8f acc = {};
    if (usepk) {
        const char* atile = (const char*)apack + (size_t)mt * 16 * 1024;
        #pragma unroll 4
        for (int kcl = 0; kcl < 16; ++kcl) {
            union { v16bf v; uint4 u[2]; } a;
            const uint4* ap = (const uint4*)(atile + (size_t)kcl * 1024 + lane * 32);
            a.u[0] = ap[0];
            a.u[1] = ap[1];
            union { v16bf v; uint4 u[2]; } b;
            const uint4* bp = (const uint4*)(wtile + (size_t)kcl * 1024 + lane * 32);
            b.u[0] = bp[0];
            b.u[1] = bp[1];
            acc = __builtin_amdgcn_wmma_f32_16x16x32_bf16(
                false, a.v, false, b.v, (short)0, acc, false, false);
        }
    } else {
        const float* rowp = afp + (size_t)(mt * 16 + (lane & 15)) * afs;
        #pragma unroll 4
        for (int kcl = 0; kcl < 16; ++kcl) {
            // raw fp32 A (X input): k = kb + (j<8 ? j : j+8), kb = kcl*32 + hi*8
            int kb = kcl * 32 + hi * 8;
            float4 f0 = *(const float4*)(rowp + kb);
            float4 f1 = *(const float4*)(rowp + kb + 4);
            float4 f2 = *(const float4*)(rowp + kb + 16);
            float4 f3 = *(const float4*)(rowp + kb + 20);
            union { v16bf v; __bf16 e[16]; } a;
            a.e[0]  = (__bf16)f0.x; a.e[1]  = (__bf16)f0.y;
            a.e[2]  = (__bf16)f0.z; a.e[3]  = (__bf16)f0.w;
            a.e[4]  = (__bf16)f1.x; a.e[5]  = (__bf16)f1.y;
            a.e[6]  = (__bf16)f1.z; a.e[7]  = (__bf16)f1.w;
            a.e[8]  = (__bf16)f2.x; a.e[9]  = (__bf16)f2.y;
            a.e[10] = (__bf16)f2.z; a.e[11] = (__bf16)f2.w;
            a.e[12] = (__bf16)f3.x; a.e[13] = (__bf16)f3.y;
            a.e[14] = (__bf16)f3.z; a.e[15] = (__bf16)f3.w;
            union { v16bf v; uint4 u[2]; } b;
            const uint4* bp = (const uint4*)(wtile + (size_t)kcl * 1024 + lane * 32);
            b.u[0] = bp[0];
            b.u[1] = bp[1];
            acc = __builtin_amdgcn_wmma_f32_16x16x32_bf16(
                false, a.v, false, b.v, (short)0, acc, false, false);
        }
    }

    // C/D layout: VGPR r -> m = r + hi*8, n = lane&15
    {
        union { v8f v; float f[8]; } ac; ac.v = acc;
        int nloc = lane & 15;
        int mb = hi * 8;
        #pragma unroll
        for (int r = 0; r < 8; ++r) lds[khalf][gate][mb + r][nloc] = ac.f[r];
    }
    __syncthreads();

    // Elementwise update: one thread per h element of the 16x16 tile
    int m = tid >> 4, n = tid & 15;
    int nc = nt * 16 + n;
    float gi = lds[0][0][m][n] + lds[1][0][m][n] + bias[0 * Hc + nc];
    float gf = lds[0][1][m][n] + lds[1][1][m][n] + bias[1 * Hc + nc];
    float gg = lds[0][2][m][n] + lds[1][2][m][n] + bias[2 * Hc + nc];
    float go = lds[0][3][m][n] + lds[1][3][m][n] + bias[3 * Hc + nc];
    float si = 1.f / (1.f + __expf(-gi));
    float sf = 1.f / (1.f + __expf(-gf));
    float so = 1.f / (1.f + __expf(-go));
    float tg = tanhf(gg);
    size_t gm = (size_t)mt * 16 + m;
    float cold = c ? c[gm * cs + nc] : 0.f;
    float c2 = sf * cold + si * tg;
    float h2 = so * tanhf(c2);
    oh[gm * ohs + nc] = h2;
    oc[gm * ocs + nc] = c2;

    // Emit bf16 A-pack of h2 for the consumer cell:
    // invert k = hi*8 + (j<8?j:j+8): r = nc&31 ->
    //   r<8: (hi=0,j=r); r<16: (1,r-8); r<24: (0,r-8); else (1,r-16)
    int r2 = nc & 31;
    int hi2 = (r2 < 8) ? 0 : (r2 < 16) ? 1 : (r2 < 24) ? 0 : 1;
    int j2  = (r2 < 8) ? r2 : (r2 < 16) ? r2 - 8 : (r2 < 24) ? r2 - 8 : r2 - 16;
    int lane2 = hi2 * 16 + m;
    ohp[(((size_t)mt * 16 + (nc >> 5)) * 32 + lane2) * 16 + j2] = (__bf16)h2;
}

// Copy final per-layer states (parity-0 buffers) into (B,L,H) output layout.
__global__ __launch_bounds__(256) void final_copy(
    const float* __restrict__ Hs, const float* __restrict__ Ms,
    float* __restrict__ H1out, float* __restrict__ M1out)
{
    int i = blockIdx.x * 256 + threadIdx.x;
    int total = Lc * Bc * Hc;
    if (i >= total) return;
    int n = i % Hc;
    int b = (i / Hc) % Bc;
    int l = i / (Hc * Bc);
    size_t s = ((size_t)l * Bc + b) * Hc + n;
    size_t d = ((size_t)b * Lc + l) * Hc + n;
    H1out[d] = Hs[s];
    M1out[d] = Ms[s];
}

extern "C" void kernel_launch(void* const* d_in, const int* in_sizes, int n_in,
                              void* d_out, int out_size, void* d_ws, size_t ws_size,
                              hipStream_t stream)
{
    const float* X   = (const float*)d_in[0];  // (B,T,H)
    const float* Wih = (const float*)d_in[1];  // (L,2,4H,H)
    const float* Whh = (const float*)d_in[2];  // (L,2,4H,H)
    const float* bih = (const float*)d_in[3];  // (L,2,4H)
    const float* bhh = (const float*)d_in[4];  // (L,2,4H)

    float* out   = (float*)d_out;
    float* H0out = out;                                    // (B,T,H)
    float* H1out = out + (size_t)Bc * Tc * Hc;             // (B,L,H)
    float* M0out = H1out + (size_t)Bc * Lc * Hc;           // (B,T,H)
    float* M1out = M0out + (size_t)Bc * Tc * Hc;           // (B,L,H)

    // Workspace (~17.7 MiB):
    //   bf16 packed weights | biases | fp32 h0/m0 | fp32 ping-pong H/M states |
    //   bf16 A-pack buffers (h0, h1, stateH[2][2])
    char*   ws     = (char*)d_ws;
    __bf16* Wp     = (__bf16*)ws;                                   // 16 MiB
    float*  biasc  = (float*)(ws + (size_t)(16 << 20));             // 32 KiB
    float*  h0buf  = biasc + 4 * NG;
    float*  m0buf  = h0buf + (size_t)Bc * Hc;
    float*  stateH = m0buf + (size_t)Bc * Hc;        // [parity][layer][B*H]
    float*  stateM = stateH + (size_t)2 * Lc * Bc * Hc;
    __bf16* h0pk   = (__bf16*)(stateM + (size_t)2 * Lc * Bc * Hc);
    __bf16* h1pk   = h0pk + APACK_HALFS;
    __bf16* stateHP = h1pk + APACK_HALFS;            // [parity][layer][APACK]
    auto Hst  = [&](int p, int l) { return stateH + ((size_t)p * Lc + l) * Bc * Hc; };
    auto Mst  = [&](int p, int l) { return stateM + ((size_t)p * Lc + l) * Bc * Hc; };
    auto HstP = [&](int p, int l) { return stateHP + ((size_t)p * Lc + l) * APACK_HALFS; };

    // Per-call prep: bf16 tile-pack weights, combine biases, zero t=0 state
    pack_weights<<<32768, 256, 0, stream>>>(Wih, Whh, Wp);
    pack_bias<<<(4 * NG + 255) / 256, 256, 0, stream>>>(bih, bhh, biasc);
    zero_f32<<<(Lc * Bc * Hc + 255) / 256, 256, 0, stream>>>(Mst(0, 0), Lc * Bc * Hc);
    zero_f32<<<((int)(Lc * APACK_HALFS / 2) + 255) / 256, 256, 0, stream>>>(
        (float*)HstP(0, 0), (int)(Lc * APACK_HALFS / 2));

    const dim3 grid(Hc / 16, Bc / 16);   // (32, 4)
    const dim3 blk(256);
    const long TS = (long)Tc * Hc;       // row stride for (B,T,H) tensors

    for (int t = 0; t < Tc; ++t) {
        int rp = t & 1, wp = rp ^ 1;
        const float* xt = X + (size_t)t * Hc;
        float* hOt = H0out + (size_t)t * Hc;
        float* mOt = M0out + (size_t)t * Hc;

        // layer 0, dir 0:  (h0,m0) = cell(x=H1[0], h=X[t], c=0)
        lstm_cell<<<grid, blk, 0, stream>>>(
            HstP(rp, 0), nullptr, 0,  nullptr, xt, TS,  nullptr, 0,
            h0buf, Hc, m0buf, Hc, h0pk, Wp + 0 * CELL_W_HALFS, biasc + 0 * NG);
        // layer 0, dir 1:  (H1[0],M1[0]) = cell(x=h0, h=H1[0], c=M1[0])
        lstm_cell<<<grid, blk, 0, stream>>>(
            h0pk, nullptr, 0,  HstP(rp, 0), nullptr, 0,  Mst(rp, 0), Hc,
            Hst(wp, 0), Hc, Mst(wp, 0), Hc, HstP(wp, 0),
            Wp + 1 * CELL_W_HALFS, biasc + 1 * NG);
        // layer 1, dir 0:  (h0,m0) = cell(x=H1[1], h=h0, c=m0) -> step outputs
        lstm_cell<<<grid, blk, 0, stream>>>(
            HstP(rp, 1), nullptr, 0,  h0pk, nullptr, 0,  m0buf, Hc,
            hOt, TS, mOt, TS, h1pk, Wp + 2 * CELL_W_HALFS, biasc + 2 * NG);
        // layer 1, dir 1:  (H1[1],M1[1]) = cell(x=h0, h=H1[1], c=M1[1])
        lstm_cell<<<grid, blk, 0, stream>>>(
            h1pk, nullptr, 0,  HstP(rp, 1), nullptr, 0,  Mst(rp, 1), Hc,
            Hst(wp, 1), Hc, Mst(wp, 1), Hc, HstP(wp, 1),
            Wp + 3 * CELL_W_HALFS, biasc + 3 * NG);
    }

    // T is even -> final states live in parity-0 buffers
    final_copy<<<(Lc * Bc * Hc + 255) / 256, 256, 0, stream>>>(
        Hst(0, 0), Mst(0, 0), H1out, M1out);
}
